// _SE3AttentionBlock_16492674417494
// MI455X (gfx1250) — matrix operand order repaired and checked by reading, not compiled, not run on hardware
//
#include <hip/hip_runtime.h>
#include <hip/hip_bf16.h>

// ---------------------------------------------------------------------------
// SE3 attention block for gfx1250 (MI455X). All GEMMs via v_wmma_f32_16x16x32_f16.
// Attention K/V LDS staging via Tensor Data Mover (tensor_load_to_lds +
// s_wait_tensorcnt) when the builtin is available.
// ---------------------------------------------------------------------------

typedef __attribute__((ext_vector_type(16))) _Float16 v16h;
typedef __attribute__((ext_vector_type(8)))  _Float16 v8h;
typedef __attribute__((ext_vector_type(8)))  float    v8f;
typedef __attribute__((ext_vector_type(4)))  unsigned int v4u;
typedef __attribute__((ext_vector_type(8)))  int v8i;
typedef __attribute__((ext_vector_type(4)))  int v4i;

#define DEV __device__ __forceinline__

static constexpr int BB   = 8;
static constexpr int NN   = 512;
static constexpr int HID  = 512;
static constexpr int HEADS= 16;
static constexpr int HD   = 32;
static constexpr int FFN  = 2048;
static constexpr int DEMB = 64;
static constexpr float NEG_BIG = -30000.0f;   // f16-representable "minus infinity"

#if defined(__has_builtin)
#if __has_builtin(__builtin_amdgcn_tensor_load_to_lds) && \
    __has_builtin(__builtin_amdgcn_s_wait_tensorcnt)
#define HAVE_TDM 1
#endif
#endif

DEV v8f wmma_f16(v16h a, v16h b, v8f c) {
  // (neg_a, A, neg_b, B, c_mod, C, reuse_a, reuse_b)
  return __builtin_amdgcn_wmma_f32_16x16x32_f16(false, a, false, b, (short)0, c, false, false);
}

#ifdef HAVE_TDM
// Issue a TDM 2-D tile load Global->LDS (CDNA5 ISA).
// Tensor: [dim1 rows][dim0 cols] elements of 2 bytes, row stride stride0 elems.
// Tile:   [tile1 rows][tile0 cols] starting at gptr; packed contiguous in LDS.
DEV void tdm_load_2d(unsigned int lds_off, const void* gptr,
                     unsigned int dim0, unsigned int dim1,
                     unsigned int tile0, unsigned int tile1,
                     unsigned int stride0) {
  const unsigned long long ga = (unsigned long long)(size_t)gptr;
  v4u g0;
  g0[0] = 1u;                                   // count=1, user descriptor
  g0[1] = lds_off;                              // lds_addr (bytes)
  g0[2] = (unsigned int)ga;                     // global_addr[31:0]
  g0[3] = ((unsigned int)(ga >> 32) & 0x01FFFFFFu) | (2u << 30);  // addr hi | type=2
  v8i g1;
  g1[0] = (int)(1u << 16);                      // workgroup_mask=0, data_size=1 (2B)
  g1[1] = (int)((dim0 & 0xFFFFu) << 16);        // tensor_dim0[15:0]
  g1[2] = (int)((dim0 >> 16) | ((dim1 & 0xFFFFu) << 16));   // dim0 hi | dim1 lo
  g1[3] = (int)((dim1 >> 16) | ((tile0 & 0xFFFFu) << 16));  // dim1 hi | tile_dim0
  g1[4] = (int)(tile1 & 0xFFFFu);               // tile_dim1 (tile_dim2 = 0)
  g1[5] = (int)stride0;                         // tensor_dim0_stride[31:0]
  g1[6] = 0;                                    // stride hi / dim1_stride lo (unused, 2D)
  g1[7] = 0;
  v4i z4 = {0, 0, 0, 0};
#if __clang_major__ >= 23
  v8i z8 = {0, 0, 0, 0, 0, 0, 0, 0};
  __builtin_amdgcn_tensor_load_to_lds(g0, g1, z4, z4, z8, 0);
#else
  __builtin_amdgcn_tensor_load_to_lds(g0, g1, z4, z4, 0);
#endif
}
#endif

// ---- fragment loaders (contiguous k order, same in A and B) ----------
// A (16x32 f16, row-major [M][K]): lane m=l&15, g=l>>4; elems0-15=K(16g..16g+15)
// contiguous
DEV v16h load_a16(const _Float16* __restrict__ A, int lda, int row0, int k0, int lane) {
  const int m = lane & 15, g = lane >> 4;
  const _Float16* p = A + (size_t)(row0 + m) * lda + k0 + 16 * g;
  v16h f;
  *(float4*)&f       = *(const float4*)p;
  *((float4*)&f + 1) = *(const float4*)(p + 8);
  return f;
}
// A fragment from f32 source (convert on the fly)
DEV v16h load_a32(const float* __restrict__ A, int lda, int row0, int k0, int lane) {
  const int m = lane & 15, g = lane >> 4;
  const float* p = A + (size_t)(row0 + m) * lda + k0 + 16 * g;
  v16h f;
#pragma unroll
  for (int e = 0; e < 8; e++) f[e] = (_Float16)p[e];
#pragma unroll
  for (int e = 0; e < 8; e++) f[8 + e] = (_Float16)p[8 + e];
  return f;
}
// B (32x16 f16, [K][N]) from N-major storage Wt[N][K]: lane n=l&15 holds
// column n, K = 16g..16g+15 contiguous
DEV v16h load_b16(const _Float16* __restrict__ Wt, int ldk, int col0, int k0, int lane) {
  const int n = lane & 15, g = lane >> 4;
  const _Float16* p = Wt + (size_t)(col0 + n) * ldk + k0 + 16 * g;
  v16h f;
  *(float4*)&f       = *(const float4*)p;
  *((float4*)&f + 1) = *(const float4*)(p + 8);
  return f;
}

// Shared GEMM core: one wave computes a 16(M) x 64(N) tile, K-loop step 32.
DEV void gemm_core(const _Float16* __restrict__ A, int lda,
                   const _Float16* __restrict__ Wt, int ldk,
                   int row0, int col0, int K, int lane, v8f acc[4]) {
  for (int k0 = 0; k0 < K; k0 += 32) {
    if (k0 + 32 < K)
      __builtin_prefetch(A + (size_t)(row0 + (lane & 15)) * lda + k0 + 32, 0, 3);
    v16h a = load_a16(A, lda, row0, k0, lane);
#pragma unroll
    for (int t = 0; t < 4; t++) {
      v16h b = load_b16(Wt, ldk, col0 + 16 * t, k0, lane);
      acc[t] = wmma_f16(a, b, acc[t]);
    }
  }
}

// ---------------------------------------------------------------------------
// Kernel 0: convert f32 weight [K][N] -> f16 transposed [N][K]
__global__ void k_cvt_transpose(const float* __restrict__ w, _Float16* __restrict__ wt,
                                int K, int N) {
  int idx = blockIdx.x * 256 + threadIdx.x;
  if (idx < K * N) {
    int k = idx / N, n = idx - k * N;
    wt[(size_t)n * K + k] = (_Float16)w[idx];
  }
}

// Kernel 1: LayerNorm(f32 row of 512) -> f16. One row per 128-thread block.
__global__ void k_layernorm(const float* __restrict__ x, const float* __restrict__ g,
                            const float* __restrict__ b, _Float16* __restrict__ out) {
  __shared__ float s1[128], s2[128];
  const int row = blockIdx.x, t = threadIdx.x;
  const float* p = x + (size_t)row * HID + t * 4;
  float4 u = *(const float4*)p;
  s1[t] = u.x + u.y + u.z + u.w;
  s2[t] = u.x * u.x + u.y * u.y + u.z * u.z + u.w * u.w;
  __syncthreads();
  for (int off = 64; off > 0; off >>= 1) {
    if (t < off) { s1[t] += s1[t + off]; s2[t] += s2[t + off]; }
    __syncthreads();
  }
  const float mu  = s1[0] * (1.0f / HID);
  const float var = s2[0] * (1.0f / HID) - mu * mu;
  const float rs  = rsqrtf(var + 1e-5f);
  const float uu[4] = {u.x, u.y, u.z, u.w};
#pragma unroll
  for (int j = 0; j < 4; j++) {
    int c = t * 4 + j;
    out[(size_t)row * HID + c] = (_Float16)((uu[j] - mu) * rs * g[c] + b[c]);
  }
}

// Kernel 2: QKV GEMM [4096,512]x[512,1536]; epilogue scatters q (pre-scaled by
// 1/sqrt(HD)) -> [b,h,i,d], k -> [b,h,j,d], v transposed -> [b,h,d,j].
__global__ void k_qkv(const _Float16* __restrict__ h, const _Float16* __restrict__ wt,
                      const float* __restrict__ bias,
                      _Float16* __restrict__ q, _Float16* __restrict__ k,
                      _Float16* __restrict__ vt) {
  const int lane = threadIdx.x & 31, wid = threadIdx.x >> 5;
  const int wave = blockIdx.x * 4 + wid;
  const int tm = wave / 24, tn = wave - tm * 24;
  const int row0 = tm * 16, col0 = tn * 64;
  v8f acc[4] = {};
  gemm_core(h, HID, wt, HID, row0, col0, HID, lane, acc);
  const int n = lane & 15, g = lane >> 4;
  const int tok0 = row0 + 8 * g;
  const int b = tok0 >> 9, i0 = tok0 & 511;
#pragma unroll
  for (int t = 0; t < 4; t++) {
    const int c0  = col0 + 16 * t;
    const int grp = c0 >> 9;          // 0=q, 1=k, 2=v
    const int f0  = c0 & 511;
    const int hh  = f0 >> 5;
    const int d   = (f0 & 31) + n;
    const float bv = bias[c0 + n];
    if (grp == 0) {
      const float qs = 0.17677669529663687f;   // 1/sqrt(32)
      _Float16* base = q + (((size_t)b * HEADS + hh) * NN + i0) * HD + d;
#pragma unroll
      for (int r = 0; r < 8; r++) base[(size_t)r * HD] = (_Float16)((acc[t][r] + bv) * qs);
    } else if (grp == 1) {
      _Float16* base = k + (((size_t)b * HEADS + hh) * NN + i0) * HD + d;
#pragma unroll
      for (int r = 0; r < 8; r++) base[(size_t)r * HD] = (_Float16)(acc[t][r] + bv);
    } else {
      v8h pv;
#pragma unroll
      for (int r = 0; r < 8; r++) pv[r] = (_Float16)(acc[t][r] + bv);
      *(v8h*)(vt + (((size_t)b * HEADS + hh) * HD + d) * NN + i0) = pv;
    }
  }
}

// Kernel 3: distance bias + mask fused. One wave per (b, i, j-tile16):
// bias[16j,16h] = df[b,i,j0:j0+16,:64] x dist_w[64,16]  (2 WMMAs), masked and
// stored transposed as biasT[b][h][j][i] (f16) for vectorized attention loads.
// NOTE: edge_mask treated as 1-byte bool array.
__global__ void k_distbias(const float* __restrict__ df, const _Float16* __restrict__ dwt,
                           const float* __restrict__ db, const unsigned char* __restrict__ mask,
                           _Float16* __restrict__ biasT) {
  const int lane = threadIdx.x & 31, wid = threadIdx.x >> 5;
  const int wave = blockIdx.x * 4 + wid;
  const int jt = wave & 31;
  const int i  = (wave >> 5) & 511;
  const int b  = wave >> 14;
  const int n = lane & 15, g = lane >> 4;
  const float* arow = df + (((size_t)b * NN + i) * NN + jt * 16) * DEMB;
  v8f acc = {};
#pragma unroll
  for (int k0 = 0; k0 < DEMB; k0 += 32) {
    v16h a  = load_a32(arow, DEMB, 0, k0, lane);
    v16h bb = load_b16(dwt, DEMB, 0, k0, lane);
    acc = wmma_f16(a, bb, acc);
  }
  const float hb = db[n];
  const unsigned char* mrow = mask + ((size_t)b * NN + i) * NN + jt * 16 + 8 * g;
  _Float16* outb = biasT + (((size_t)b * HEADS + n) * NN) * NN + i;
#pragma unroll
  for (int r = 0; r < 8; r++) {
    float v = acc[r] + hb;
    if (mrow[r] == 0) v = NEG_BIG;
    const int j = jt * 16 + 8 * g + r;
    outb[(size_t)j * NN] = (_Float16)v;
  }
}

// Kernel 4: flash attention. Block = 4 waves; wave w owns i-tile i0..i0+15 of
// one (b,h). K/V 32-blocks staged into LDS by the Tensor Data Mover (one
// descriptor pair per workgroup iteration, issued by wave 0, synced with
// s_wait_tensorcnt + barrier); P staged via LDS to convert C-layout->A-layout.
__global__ void k_attn(const _Float16* __restrict__ q, const _Float16* __restrict__ kk,
                       const _Float16* __restrict__ vt, const _Float16* __restrict__ biasT,
                       _Float16* __restrict__ ctx) {
  __shared__ _Float16 Kblk[32 * 32];     // [j_local][d]
  __shared__ _Float16 Vblk[32 * 32];     // [d][j_local]  (v stored transposed)
  __shared__ _Float16 Pbuf[4][16 * 32];  // per-wave [i_local][j_local]
  const int bh   = blockIdx.x >> 3;      // b*16 + h
  const int iblk = blockIdx.x & 7;
  const int wid = threadIdx.x >> 5, lane = threadIdx.x & 31;
  const int n = lane & 15, g = lane >> 4;
  const int i0 = iblk * 64 + wid * 16;

  const _Float16* qbase = q  + ((size_t)bh * NN + i0) * HD;
  const _Float16* kbase = kk + (size_t)bh * NN * HD;
  const _Float16* vbase = vt + (size_t)bh * HD * NN;
  const _Float16* bbase = biasT + (size_t)bh * NN * NN;

  // q A-fragment (held for the whole kernel; K dim = HD = 32 exactly)
  v16h qa;
  {
    const _Float16* p = qbase + (size_t)n * HD + 16 * g;
    *(float4*)&qa       = *(const float4*)p;
    *((float4*)&qa + 1) = *(const float4*)(p + 8);
  }

  v8f O0 = {}, O1 = {};
  float mrow[8], lrow[8];
#pragma unroll
  for (int r = 0; r < 8; r++) { mrow[r] = -3.0e38f; lrow[r] = 0.0f; }

  for (int jb = 0; jb < NN; jb += 32) {
    __syncthreads();  // previous-iteration LDS readers done before overwrite
#ifdef HAVE_TDM
    if (wid == 0) {
      // K tile: tensor [512 j][32 d], tile [32 j][32 d] at row jb
      tdm_load_2d((unsigned int)(size_t)&Kblk[0], kbase + (size_t)jb * HD,
                  HD, NN, 32, 32, HD);
      // V tile: tensor [32 d][512 j], tile [32 d][32 j] at col jb
      tdm_load_2d((unsigned int)(size_t)&Vblk[0], vbase + jb,
                  NN, HD, 32, 32, NN);
      __builtin_amdgcn_s_wait_tensorcnt((short)0);
    }
#else
    {
      const int row = threadIdx.x >> 2, ch = threadIdx.x & 3;
      *(float4*)&Kblk[row * 32 + ch * 8] = *(const float4*)&kbase[(size_t)(jb + row) * HD + ch * 8];
      *(float4*)&Vblk[row * 32 + ch * 8] = *(const float4*)&vbase[(size_t)row * NN + jb + ch * 8];
    }
#endif
    __syncthreads();

    // S = q x k^T for two 16-wide j sub-tiles
    v16h kb0, kb1;
    {
      const _Float16* p0 = &Kblk[n * 32 + 16 * g];
      *(float4*)&kb0       = *(const float4*)p0;
      *((float4*)&kb0 + 1) = *(const float4*)(p0 + 8);
      const _Float16* p1 = &Kblk[(16 + n) * 32 + 16 * g];
      *(float4*)&kb1       = *(const float4*)p1;
      *((float4*)&kb1 + 1) = *(const float4*)(p1 + 8);
    }
    v8f S0 = {}, S1 = {};
    S0 = wmma_f16(qa, kb0, S0);
    S1 = wmma_f16(qa, kb1, S1);

    // + bias(+mask) : biasT[b][h][j][i], i contiguous -> 16B vector per lane
    {
      v8h b0 = *(const v8h*)&bbase[(size_t)(jb + n) * NN + i0 + 8 * g];
      v8h b1 = *(const v8h*)&bbase[(size_t)(jb + 16 + n) * NN + i0 + 8 * g];
#pragma unroll
      for (int r = 0; r < 8; r++) { S0[r] += (float)b0[r]; S1[r] += (float)b1[r]; }
    }

    // online softmax (row = i lives at (g, r); reduce over the 16 n-lanes)
    float tm[8], al[8], ps[8];
#pragma unroll
    for (int r = 0; r < 8; r++) tm[r] = fmaxf(S0[r], S1[r]);
#pragma unroll
    for (int off = 1; off < 16; off <<= 1)
#pragma unroll
      for (int r = 0; r < 8; r++) tm[r] = fmaxf(tm[r], __shfl_xor(tm[r], off, 32));
#pragma unroll
    for (int r = 0; r < 8; r++) {
      const float mn = fmaxf(mrow[r], tm[r]);
      al[r] = __expf(mrow[r] - mn);
      mrow[r] = mn;
      const float p0 = __expf(S0[r] - mn);
      const float p1 = __expf(S1[r] - mn);
      ps[r] = p0 + p1;
      O0[r] *= al[r];
      O1[r] *= al[r];
      Pbuf[wid][(8 * g + r) * 32 + n]      = (_Float16)p0;
      Pbuf[wid][(8 * g + r) * 32 + 16 + n] = (_Float16)p1;
    }
#pragma unroll
    for (int off = 1; off < 16; off <<= 1)
#pragma unroll
      for (int r = 0; r < 8; r++) ps[r] += __shfl_xor(ps[r], off, 32);
#pragma unroll
    for (int r = 0; r < 8; r++) lrow[r] = lrow[r] * al[r] + ps[r];

    __syncthreads();  // P visible to all lanes of this wave

    // O += P x V  (A = P 16x32; B = V[32j x 16d] x2)
    v16h pa, vb0, vb1;
    {
      const _Float16* p = &Pbuf[wid][n * 32 + 16 * g];
      *(float4*)&pa       = *(const float4*)p;
      *((float4*)&pa + 1) = *(const float4*)(p + 8);
      const _Float16* v0 = &Vblk[n * 32 + 16 * g];
      *(float4*)&vb0       = *(const float4*)v0;
      *((float4*)&vb0 + 1) = *(const float4*)(v0 + 8);
      const _Float16* v1 = &Vblk[(16 + n) * 32 + 16 * g];
      *(float4*)&vb1       = *(const float4*)v1;
      *((float4*)&vb1 + 1) = *(const float4*)(v1 + 8);
    }
    O0 = wmma_f16(pa, vb0, O0);
    O1 = wmma_f16(pa, vb1, O1);
  }

  // write ctx[token][h*32+d] (f16)
  const int b = bh >> 4, hh = bh & 15;
#pragma unroll
  for (int r = 0; r < 8; r++) {
    const float inv = 1.0f / lrow[r];
    const int i = i0 + 8 * g + r;
    const size_t tok = (size_t)b * NN + i;
    ctx[tok * HID + hh * HD + n]      = (_Float16)(O0[r] * inv);
    ctx[tok * HID + hh * HD + 16 + n] = (_Float16)(O1[r] * inv);
  }
}

// Kernel 5: out-projection + residual -> x2 (f32)
__global__ void k_outproj(const _Float16* __restrict__ ctx, const _Float16* __restrict__ wt,
                          const float* __restrict__ ob, const float* __restrict__ x,
                          float* __restrict__ x2) {
  const int lane = threadIdx.x & 31, wid = threadIdx.x >> 5;
  const int wave = blockIdx.x * 4 + wid;
  const int tm = wave >> 3, tn = wave & 7;
  const int row0 = tm * 16, col0 = tn * 64;
  v8f acc[4] = {};
  gemm_core(ctx, HID, wt, HID, row0, col0, HID, lane, acc);
  const int n = lane & 15, g = lane >> 4;
#pragma unroll
  for (int t = 0; t < 4; t++) {
    const int c = col0 + 16 * t + n;
    const float bv = ob[c];
#pragma unroll
    for (int r = 0; r < 8; r++) {
      const size_t idx = (size_t)(row0 + 8 * g + r) * HID + c;
      x2[idx] = acc[t][r] + bv + x[idx];
    }
  }
}

// Kernel 6: FF1 + exact GeLU -> f16 mid
__global__ void k_ff1(const _Float16* __restrict__ y, const _Float16* __restrict__ wt,
                      const float* __restrict__ fb, _Float16* __restrict__ mid) {
  const int lane = threadIdx.x & 31, wid = threadIdx.x >> 5;
  const int wave = blockIdx.x * 4 + wid;
  const int tm = wave >> 5, tn = wave & 31;
  const int row0 = tm * 16, col0 = tn * 64;
  v8f acc[4] = {};
  gemm_core(y, HID, wt, HID, row0, col0, HID, lane, acc);
  const int n = lane & 15, g = lane >> 4;
#pragma unroll
  for (int t = 0; t < 4; t++) {
    const int c = col0 + 16 * t + n;
    const float bv = fb[c];
#pragma unroll
    for (int r = 0; r < 8; r++) {
      const float z = acc[t][r] + bv;
      const float ge = 0.5f * z * (1.0f + erff(z * 0.70710678118654752f));
      mid[(size_t)(row0 + 8 * g + r) * FFN + c] = (_Float16)ge;
    }
  }
}

// Kernel 7: FF2 + residual -> final output (f32)
__global__ void k_ff2(const _Float16* __restrict__ mid, const _Float16* __restrict__ wt,
                      const float* __restrict__ fb, const float* __restrict__ x2,
                      float* __restrict__ out) {
  const int lane = threadIdx.x & 31, wid = threadIdx.x >> 5;
  const int wave = blockIdx.x * 4 + wid;
  const int tm = wave >> 3, tn = wave & 7;
  const int row0 = tm * 16, col0 = tn * 64;
  v8f acc[4] = {};
  gemm_core(mid, FFN, wt, FFN, row0, col0, FFN, lane, acc);
  const int n = lane & 15, g = lane >> 4;
#pragma unroll
  for (int t = 0; t < 4; t++) {
    const int c = col0 + 16 * t + n;
    const float bv = fb[c];
#pragma unroll
    for (int r = 0; r < 8; r++) {
      const size_t idx = (size_t)(row0 + 8 * g + r) * HID + c;
      out[idx] = acc[t][r] + bv + x2[idx];
    }
  }
}

// ---------------------------------------------------------------------------
extern "C" void kernel_launch(void* const* d_in, const int* in_sizes, int n_in,
                              void* d_out, int out_size, void* d_ws, size_t ws_size,
                              hipStream_t stream) {
  (void)in_sizes; (void)n_in; (void)out_size; (void)ws_size;
  const float* x      = (const float*)d_in[0];
  const float* df     = (const float*)d_in[1];
  const unsigned char* mask = (const unsigned char*)d_in[2];  // bool array
  const float* qkv_w  = (const float*)d_in[3];
  const float* qkv_b  = (const float*)d_in[4];
  const float* out_w  = (const float*)d_in[5];
  const float* out_b  = (const float*)d_in[6];
  const float* dist_w = (const float*)d_in[7];
  const float* dist_b = (const float*)d_in[8];
  const float* ln1_g  = (const float*)d_in[9];
  const float* ln1_b  = (const float*)d_in[10];
  const float* ln2_g  = (const float*)d_in[11];
  const float* ln2_b  = (const float*)d_in[12];
  const float* ff1_w  = (const float*)d_in[13];
  const float* ff1_b  = (const float*)d_in[14];
  const float* ff2_w  = (const float*)d_in[15];
  const float* ff2_b  = (const float*)d_in[16];
  float* outp = (float*)d_out;

  char* ws = (char*)d_ws;
  size_t off = 0;
  auto take = [&](size_t bytes) -> char* {
    char* p = ws + off;
    off = (off + bytes + 255) & ~(size_t)255;
    return p;
  };
  const size_t TOK = (size_t)BB * NN;                 // 4096
  _Float16* h16    = (_Float16*)take(TOK * HID * 2);
  _Float16* y16    = (_Float16*)take(TOK * HID * 2);
  _Float16* q16    = (_Float16*)take((size_t)BB * HEADS * NN * HD * 2);
  _Float16* k16    = (_Float16*)take((size_t)BB * HEADS * NN * HD * 2);
  _Float16* vt16   = (_Float16*)take((size_t)BB * HEADS * HD * NN * 2);
  _Float16* ctx16  = (_Float16*)take(TOK * HID * 2);
  _Float16* mid16  = (_Float16*)take(TOK * FFN * 2);
  float*    x2     = (float*)   take(TOK * HID * 4);
  _Float16* biasT  = (_Float16*)take((size_t)BB * HEADS * NN * NN * 2);  // 64 MB
  _Float16* qkv_wt = (_Float16*)take((size_t)HID * 3 * HID * 2);
  _Float16* out_wt = (_Float16*)take((size_t)HID * HID * 2);
  _Float16* ff1_wt = (_Float16*)take((size_t)HID * FFN * 2);
  _Float16* ff2_wt = (_Float16*)take((size_t)FFN * HID * 2);
  _Float16* dist_wt= (_Float16*)take((size_t)DEMB * HEADS * 2);

  const dim3 blk128(128), blk256(256);

  // weight conversion / transposition (f32 [K][N] -> f16 [N][K])
  k_cvt_transpose<<<(HID * 3 * HID + 255) / 256, blk256, 0, stream>>>(qkv_w, qkv_wt, HID, 3 * HID);
  k_cvt_transpose<<<(HID * HID + 255) / 256,     blk256, 0, stream>>>(out_w, out_wt, HID, HID);
  k_cvt_transpose<<<(HID * FFN + 255) / 256,     blk256, 0, stream>>>(ff1_w, ff1_wt, HID, FFN);
  k_cvt_transpose<<<(FFN * HID + 255) / 256,     blk256, 0, stream>>>(ff2_w, ff2_wt, FFN, HID);
  k_cvt_transpose<<<(DEMB * HEADS + 255) / 256,  blk256, 0, stream>>>(dist_w, dist_wt, DEMB, HEADS);

  // LN1 -> h16
  k_layernorm<<<(int)TOK, blk128, 0, stream>>>(x, ln1_g, ln1_b, h16);
  // QKV projection (waves: 256 m-tiles x 24 n-tiles = 6144; 4 waves/block)
  k_qkv<<<6144 / 4, blk128, 0, stream>>>(h16, qkv_wt, qkv_b, q16, k16, vt16);
  // distance bias + mask (waves: 8 * 512 * 32 = 131072)
  k_distbias<<<131072 / 4, blk128, 0, stream>>>(df, dist_wt, dist_b, mask, biasT);
  // flash attention: 128 (b,h) x 8 i-blocks
  k_attn<<<128 * 8, blk128, 0, stream>>>(q16, k16, vt16, biasT, ctx16);
  // out-projection + residual (waves: 256 x 8 = 2048)
  k_outproj<<<2048 / 4, blk128, 0, stream>>>(ctx16, out_wt, out_b, x, x2);
  // LN2 -> y16
  k_layernorm<<<(int)TOK, blk128, 0, stream>>>(x2, ln2_g, ln2_b, y16);
  // FF1 + GeLU (waves: 256 x 32 = 8192)
  k_ff1<<<8192 / 4, blk128, 0, stream>>>(y16, ff1_wt, ff1_b, mid16);
  // FF2 + residual -> output (waves: 256 x 8 = 2048)
  k_ff2<<<2048 / 4, blk128, 0, stream>>>(mid16, ff2_wt, ff2_b, x2, outp);
}
